// DeepSpeedBlockBaselineLayer_15530601742502
// MI455X (gfx1250) — matrix operand order repaired and checked by reading, not compiled, not run on hardware
//
#include <hip/hip_runtime.h>
#include <hip/hip_bf16.h>

// ---------------------------------------------------------------------------
// MI455X (gfx1250) implementation. Wave32, WMMA bf16 (16x16x32, f32 accum).
// HBM-bound on fp32 weight reads (~350MB -> ~15us floor @ 23.3TB/s), so all
// GEMMs read fp32 weights once (coalesced, async-to-LDS where available),
// convert to bf16, and run the v_wmma_f32_16x16x32_bf16 pipe.
// ---------------------------------------------------------------------------

typedef __attribute__((ext_vector_type(16))) __bf16 v16bf;
typedef __attribute__((ext_vector_type(8)))  __bf16 v8bf;
typedef __attribute__((ext_vector_type(8)))  float  v8f;
typedef __attribute__((__vector_size__(4 * sizeof(int)))) int b128_t;

#define S_LEN   2048
#define BATCH   2
#define HIDDEN  1024
#define NTOK    (S_LEN * BATCH)   // 4096
#define NHEADS  16
#define NKV     8
#define HEAD_DIM 64
#define QKV_OUT 2048
#define NEXP    8
#define FFN     4096
#define CAP     1280              // ceil(4096*2/8*1.25)

#if __has_builtin(__builtin_amdgcn_global_load_async_to_lds_b128)
#define USE_ASYNC_LDS 1
#endif

static __device__ inline void wait_async0() {
#if __has_builtin(__builtin_amdgcn_s_wait_asynccnt)
  __builtin_amdgcn_s_wait_asynccnt(0);
#else
  asm volatile("s_wait_asynccnt 0x0" ::: "memory");
#endif
}

// Copy 16 bytes global->LDS. Async (ASYNCcnt-tracked, VGPR-bypassing) when
// the toolchain exposes the gfx1250 builtin; synchronous fallback otherwise.
static __device__ inline void cp16_g2l(const float* __restrict__ g,
                                       float* __restrict__ l) {
#ifdef USE_ASYNC_LDS
  __builtin_amdgcn_global_load_async_to_lds_b128((b128_t*)g, (b128_t*)l, 0, 0);
#else
  l[0] = g[0];
  l[1] = g[1];
  l[2] = g[2];
  l[3] = g[3];
#endif
}

static __device__ inline void cp_tile_fence() {
#ifdef USE_ASYNC_LDS
  wait_async0();
#endif
}

static __device__ inline v8f zero_v8f() {
  v8f z = {0.f, 0.f, 0.f, 0.f, 0.f, 0.f, 0.f, 0.f};
  return z;
}

static __device__ inline v8f wmma_bf16(v16bf a, v16bf b, v8f c) {
  return __builtin_amdgcn_wmma_f32_16x16x32_bf16(false, a, false, b,
                                                 (short)0, c, false, false);
}

// A-fragment (16x32 bf16) per ISA layout: lane<16 holds row, K {0..7,16..23};
// lane>=16 holds K {8..15,24..31}. Caller passes row-ptr + (hi*8) base.
static __device__ inline v16bf load_a_bf16(const __bf16* p) {
  v8bf lo = *reinterpret_cast<const v8bf*>(p);
  v8bf hi = *reinterpret_cast<const v8bf*>(p + 16);
  return __builtin_shufflevector(lo, hi, 0, 1, 2, 3, 4, 5, 6, 7,
                                 8, 9, 10, 11, 12, 13, 14, 15);
}

static __device__ inline float gelu_f(float x) {
  return 0.5f * x * (1.f + erff(x * 0.70710678118654752f));
}

// --------------------------- LayerNorm -> bf16 -----------------------------
__global__ __launch_bounds__(256) void ln_bf16_kernel(
    const float* __restrict__ x, const float* __restrict__ w,
    const float* __restrict__ b, __bf16* __restrict__ y) {
  __shared__ float red[256];
  int t = blockIdx.x;
  const float* xr = x + (size_t)t * HIDDEN;
  float v[4];
  float s = 0.f;
#pragma unroll
  for (int i = 0; i < 4; i++) {
    v[i] = xr[threadIdx.x + i * 256];
    s += v[i];
  }
  red[threadIdx.x] = s;
  __syncthreads();
  for (int o = 128; o; o >>= 1) {
    if (threadIdx.x < o) red[threadIdx.x] += red[threadIdx.x + o];
    __syncthreads();
  }
  float mu = red[0] * (1.f / HIDDEN);
  __syncthreads();
  float s2 = 0.f;
#pragma unroll
  for (int i = 0; i < 4; i++) {
    float d = v[i] - mu;
    s2 += d * d;
  }
  red[threadIdx.x] = s2;
  __syncthreads();
  for (int o = 128; o; o >>= 1) {
    if (threadIdx.x < o) red[threadIdx.x] += red[threadIdx.x + o];
    __syncthreads();
  }
  float rstd = rsqrtf(red[0] * (1.f / HIDDEN) + 1e-5f);
  __bf16* yr = y + (size_t)t * HIDDEN;
#pragma unroll
  for (int i = 0; i < 4; i++) {
    int h = threadIdx.x + i * 256;
    yr[h] = (__bf16)((v[i] - mu) * rstd * w[h] + b[h]);
  }
}

// --------------- GEMM: C[M,N] = A[M,K](bf16) x W[N,K](fp32)^T --------------
// 8 waves/block, wave tile 32x32 (4 WMMA accumulators), block tile 64x128.
// W tile (128 rows x 32 K) staged fp32 through LDS with coalesced row loads
// (async-to-LDS on gfx1250). EPI==1 fuses residual add.
#define WT_LDB 36  // 32 K + pad, keeps 16B-aligned b128 chunks, breaks banks
template <int EPI>
__global__ __launch_bounds__(256) void gemm_wt_kernel(
    const __bf16* __restrict__ A, const float* __restrict__ W,
    float* __restrict__ C, const float* __restrict__ R, int N, int K) {
  __shared__ float Ws[128 * WT_LDB];
  int lane = threadIdx.x & 31;
  int wv = threadIdx.x >> 5;
  int lo = lane & 15, hi = lane >> 4;
  int m0 = blockIdx.x * 64 + (wv >> 2) * 32;
  int nblk = blockIdx.y * 128;
  int n0 = nblk + (wv & 3) * 32;
  int lrow0 = (wv & 3) * 32 + lo;  // local W row for b0 fragment

  v8f c00 = zero_v8f(), c01 = zero_v8f(), c10 = zero_v8f(), c11 = zero_v8f();
  const __bf16* a0p = A + (size_t)(m0 + lo) * K + hi * 8;
  const __bf16* a1p = a0p + (size_t)16 * K;

  for (int k0 = 0; k0 < K; k0 += 32) {
    // stage W[nblk..nblk+127][k0..k0+31]: 1024 x 16B chunks, 4 per thread
#pragma unroll
    for (int i = 0; i < 4; i++) {
      int ch = threadIdx.x + i * 256;
      int r = ch >> 3;           // 0..127 (8 chunks per 32-float row)
      int cf = (ch & 7) * 4;     // 0..28
      cp16_g2l(W + (size_t)(nblk + r) * K + k0 + cf, Ws + r * WT_LDB + cf);
    }
    cp_tile_fence();
    __syncthreads();

    v16bf a0 = load_a_bf16(a0p + k0);
    v16bf a1 = load_a_bf16(a1p + k0);
    v16bf b0, b1;
#pragma unroll
    for (int i = 0; i < 16; i++) {
      int kc = hi * 8 + i + (i & 8);
      b0[i] = (__bf16)Ws[lrow0 * WT_LDB + kc];
      b1[i] = (__bf16)Ws[(lrow0 + 16) * WT_LDB + kc];
    }
    c00 = wmma_bf16(a0, b0, c00);
    c01 = wmma_bf16(a0, b1, c01);
    c10 = wmma_bf16(a1, b0, c10);
    c11 = wmma_bf16(a1, b1, c11);
    __syncthreads();
  }

  int rb = m0 + hi * 8;
  int col0 = n0 + lo, col1 = col0 + 16;
#pragma unroll
  for (int j = 0; j < 8; j++) {
    size_t r0 = (size_t)(rb + j) * N, r1 = (size_t)(rb + j + 16) * N;
    float v00 = c00[j], v01 = c01[j], v10 = c10[j], v11 = c11[j];
    if (EPI == 1) {
      v00 += R[r0 + col0];
      v01 += R[r0 + col1];
      v10 += R[r1 + col0];
      v11 += R[r1 + col1];
    }
    C[r0 + col0] = v00;
    C[r0 + col1] = v01;
    C[r1 + col0] = v10;
    C[r1 + col1] = v11;
  }
}

// ------------------------- Flash attention (GQA) ---------------------------
// One wave per (b, head, 16 query rows). Online softmax, key tiles of 32.
__global__ __launch_bounds__(32) void flash_attn_kernel(
    const float* __restrict__ qkv, __bf16* __restrict__ attn_out) {
  __shared__ __bf16 P[16 * 32];  // C-layout -> A-layout relay for P
  int lane = threadIdx.x;
  int lo = lane & 15, hi = lane >> 4;
  int qtile = blockIdx.x & (S_LEN / 16 - 1);
  int bh = blockIdx.x >> 7;
  int b = bh & (BATCH - 1);
  int h = bh >> 1;
  int g = h >> 1, jq = h & 1;  // kv group, q-in-group
  int qoff = g * 256 + jq * 64;
  int koff = g * 256 + 128;
  int voff = g * 256 + 192;
  int q0 = qtile * 16;

  // Q fragments (K = 64 -> 2 A-frags), pre-scaled by 1/sqrt(64)
  v16bf qa[2];
  {
    const float* qp = qkv + ((size_t)(q0 + lo) * BATCH + b) * QKV_OUT + qoff;
#pragma unroll
    for (int f = 0; f < 2; f++) {
      int kb = f * 32 + hi * 16;
#pragma unroll
      for (int i = 0; i < 8; i++) qa[f][i] = (__bf16)(qp[kb + i] * 0.125f);
#pragma unroll
      for (int i = 0; i < 8; i++) qa[f][8 + i] = (__bf16)(qp[kb + 8 + i] * 0.125f);
    }
  }

  v8f o0 = zero_v8f(), o1 = zero_v8f(), o2 = zero_v8f(), o3 = zero_v8f();
  float mrun[8], lrun[8];
#pragma unroll
  for (int j = 0; j < 8; j++) {
    mrun[j] = -3.0e38f;
    lrun[j] = 0.f;
  }

  int qmax = q0 + 15;
  for (int kb0 = 0; kb0 <= qmax; kb0 += 32) {
    // scores S[16,32] = Q x K^T (two 16x16 column tiles)
    v8f s0 = zero_v8f(), s1 = zero_v8f();
    {
      const float* kp0 = qkv + ((size_t)(kb0 + lo) * BATCH + b) * QKV_OUT + koff;
      const float* kp1 = qkv + ((size_t)(kb0 + 16 + lo) * BATCH + b) * QKV_OUT + koff;
#pragma unroll
      for (int df = 0; df < 2; df++) {
        int dbase = df * 32 + hi * 16;
        v16bf k0f, k1f;
#pragma unroll
        for (int i = 0; i < 16; i++) {
          k0f[i] = (__bf16)kp0[dbase + i];
          k1f[i] = (__bf16)kp1[dbase + i];
        }
        s0 = wmma_bf16(qa[df], k0f, s0);
        s1 = wmma_bf16(qa[df], k1f, s1);
      }
    }
    // causal mask + online softmax statistics (row = C-frag vgpr index)
    int key0 = kb0 + lo, key1 = kb0 + 16 + lo;
    float e0[8], e1[8];
#pragma unroll
    for (int j = 0; j < 8; j++) {
      int row = q0 + j + hi * 8;
      float v0 = (key0 <= row) ? s0[j] : -1.0e30f;
      float v1 = (key1 <= row) ? s1[j] : -1.0e30f;
      float mx = fmaxf(v0, v1);
#pragma unroll
      for (int off = 8; off; off >>= 1) mx = fmaxf(mx, __shfl_xor(mx, off, 32));
      float mnew = fmaxf(mrun[j], mx);
      float sc = __expf(mrun[j] - mnew);
      float p0 = __expf(v0 - mnew);
      float p1 = __expf(v1 - mnew);
      float rs = p0 + p1;
#pragma unroll
      for (int off = 8; off; off >>= 1) rs += __shfl_xor(rs, off, 32);
      lrun[j] = lrun[j] * sc + rs;
      mrun[j] = mnew;
      o0[j] *= sc;
      o1[j] *= sc;
      o2[j] *= sc;
      o3[j] *= sc;
      e0[j] = p0;
      e1[j] = p1;
    }
    __syncthreads();
#pragma unroll
    for (int j = 0; j < 8; j++) {
      int m = j + hi * 8;
      P[m * 32 + lo] = (__bf16)e0[j];
      P[m * 32 + 16 + lo] = (__bf16)e1[j];
    }
    __syncthreads();
    // reload P as A-fragment (16x32)
    v16bf pa;
    {
      const __bf16* pr = P + lo * 32 + hi * 16;
#pragma unroll
      for (int i = 0; i < 8; i++) pa[i] = pr[i];
#pragma unroll
      for (int i = 0; i < 8; i++) pa[8 + i] = pr[8 + i];
    }
    // out += P x V   (V tile 32x64 -> 4 B-frags)
    {
      const float* vbase =
          qkv + ((size_t)(kb0 + hi * 16) * BATCH + b) * QKV_OUT + voff;
#pragma unroll
      for (int nt = 0; nt < 4; nt++) {
        v16bf vf;
        int d = nt * 16 + lo;
#pragma unroll
        for (int i = 0; i < 16; i++)
          vf[i] = (__bf16)vbase[(size_t)i * BATCH * QKV_OUT + d];
        if (nt == 0) o0 = wmma_bf16(pa, vf, o0);
        else if (nt == 1) o1 = wmma_bf16(pa, vf, o1);
        else if (nt == 2) o2 = wmma_bf16(pa, vf, o2);
        else o3 = wmma_bf16(pa, vf, o3);
      }
    }
  }
#pragma unroll
  for (int j = 0; j < 8; j++) {
    int s = q0 + j + hi * 8;
    size_t ro = ((size_t)s * BATCH + b) * HIDDEN + (size_t)h * HEAD_DIM;
    float inv = 1.f / lrun[j];
    attn_out[ro + lo] = (__bf16)(o0[j] * inv);
    attn_out[ro + 16 + lo] = (__bf16)(o1[j] * inv);
    attn_out[ro + 32 + lo] = (__bf16)(o2[j] * inv);
    attn_out[ro + 48 + lo] = (__bf16)(o3[j] * inv);
  }
}

// ------------------------------- Router ------------------------------------
__global__ __launch_bounds__(256) void router_kernel(
    const __bf16* __restrict__ x, const float* __restrict__ Wr,
    float* __restrict__ top_p, int* __restrict__ top_i) {
  int lane = threadIdx.x & 31, wv = threadIdx.x >> 5;
  int t = blockIdx.x * 8 + wv;
  float acc[8];
#pragma unroll
  for (int e = 0; e < 8; e++) acc[e] = 0.f;
  const __bf16* xr = x + (size_t)t * HIDDEN + lane * 32;
  const float* wr0 = Wr + (size_t)lane * 32 * NEXP;
  for (int hh = 0; hh < 32; hh++) {
    float xv = (float)xr[hh];
    const float* wr = wr0 + (size_t)hh * NEXP;
#pragma unroll
    for (int e = 0; e < 8; e++) acc[e] += xv * wr[e];
  }
#pragma unroll
  for (int off = 16; off; off >>= 1) {
#pragma unroll
    for (int e = 0; e < 8; e++) acc[e] += __shfl_xor(acc[e], off, 32);
  }
  if (lane == 0) {
    int i0 = 0;
    float m0v = acc[0];
#pragma unroll
    for (int e = 1; e < 8; e++)
      if (acc[e] > m0v) { m0v = acc[e]; i0 = e; }
    int i1 = -1;
    float m1v = -3e38f;
#pragma unroll
    for (int e = 0; e < 8; e++)
      if (e != i0 && acc[e] > m1v) { m1v = acc[e]; i1 = e; }
    float e1 = __expf(m1v - m0v);
    float denom = 1.f + e1;
    top_i[t * 2] = i0;
    top_i[t * 2 + 1] = i1;
    top_p[t * 2] = 1.f / denom;
    top_p[t * 2 + 1] = e1 / denom;
  }
}

// Deterministic stable ranks (reproduces jnp.argsort stable-by-expert).
__global__ void rank_kernel(const int* __restrict__ flat_exp,
                            int* __restrict__ rank) {
  int e = threadIdx.x;
  if (e >= NEXP) return;
  int cnt = 0;
  for (int i = 0; i < NTOK * 2; i++) {
    if (flat_exp[i] == e) rank[i] = cnt++;
  }
}

__global__ __launch_bounds__(256) void dispatch_kernel(
    const __bf16* __restrict__ ln2, const int* __restrict__ flat_exp,
    const int* __restrict__ rank, __bf16* __restrict__ exp_in) {
  int i = blockIdx.x;  // flat entry (token*2 + j)
  int r = rank[i];
  if (r >= CAP) return;  // capacity-dropped
  int e = flat_exp[i];
  size_t src = (size_t)(i >> 1) * HIDDEN;
  size_t dst = ((size_t)e * CAP + r) * HIDDEN;
#pragma unroll
  for (int c = 0; c < 4; c++)
    exp_in[dst + threadIdx.x + c * 256] = ln2[src + threadIdx.x + c * 256];
}

// -------- Grouped MoE GEMM: C = A[M,K](bf16) x B[K,N](fp32), per expert ----
// B tile (32 K x 128 N) staged fp32 through LDS (async-to-LDS on gfx1250).
// EPI==0: GELU -> bf16 store; EPI==1: fp32 store.
#define NT_LDB 132  // 128 N + pad (keeps 16B chunks aligned)
template <int EPI>
__global__ __launch_bounds__(256) void gemm_moe_kernel(
    const __bf16* __restrict__ Abase, const float* __restrict__ Bbase,
    void* __restrict__ Cbase, int N, int K, size_t strideA, size_t strideB,
    size_t strideC) {
  __shared__ float Bs[32 * NT_LDB];
  int e = blockIdx.z;
  const __bf16* A = Abase + (size_t)e * strideA;
  const float* B = Bbase + (size_t)e * strideB;

  int lane = threadIdx.x & 31, wv = threadIdx.x >> 5;
  int lo = lane & 15, hi = lane >> 4;
  int m0 = blockIdx.x * 64 + (wv >> 2) * 32;
  int nb = blockIdx.y * 128;
  int n0 = nb + (wv & 3) * 32;
  int cl0 = (wv & 3) * 32 + lo;

  v8f c00 = zero_v8f(), c01 = zero_v8f(), c10 = zero_v8f(), c11 = zero_v8f();
  const __bf16* a0p = A + (size_t)(m0 + lo) * K + hi * 8;
  const __bf16* a1p = a0p + (size_t)16 * K;

  for (int k0 = 0; k0 < K; k0 += 32) {
    // stage B[k0..k0+31][nb..nb+127]: 1024 x 16B chunks, 4 per thread
#pragma unroll
    for (int i = 0; i < 4; i++) {
      int ch = threadIdx.x + i * 256;
      int r = ch >> 5;           // 0..31 (32 chunks per 128-float row)
      int cf = (ch & 31) * 4;    // 0..124
      cp16_g2l(B + (size_t)(k0 + r) * N + nb + cf, Bs + r * NT_LDB + cf);
    }
    cp_tile_fence();
    __syncthreads();

    v16bf a0 = load_a_bf16(a0p + k0);
    v16bf a1 = load_a_bf16(a1p + k0);
    v16bf b0, b1;
#pragma unroll
    for (int i = 0; i < 16; i++) {
      int kr = hi * 8 + i + (i & 8);
      b0[i] = (__bf16)Bs[kr * NT_LDB + cl0];
      b1[i] = (__bf16)Bs[kr * NT_LDB + cl0 + 16];
    }
    c00 = wmma_bf16(a0, b0, c00);
    c01 = wmma_bf16(a0, b1, c01);
    c10 = wmma_bf16(a1, b0, c10);
    c11 = wmma_bf16(a1, b1, c11);
    __syncthreads();
  }

  int rb = m0 + hi * 8;
  int col0 = n0 + lo, col1 = col0 + 16;
  if (EPI == 0) {
    __bf16* C = (__bf16*)Cbase + (size_t)e * strideC;
#pragma unroll
    for (int j = 0; j < 8; j++) {
      size_t r0 = (size_t)(rb + j) * N, r1 = (size_t)(rb + j + 16) * N;
      C[r0 + col0] = (__bf16)gelu_f(c00[j]);
      C[r0 + col1] = (__bf16)gelu_f(c01[j]);
      C[r1 + col0] = (__bf16)gelu_f(c10[j]);
      C[r1 + col1] = (__bf16)gelu_f(c11[j]);
    }
  } else {
    float* C = (float*)Cbase + (size_t)e * strideC;
#pragma unroll
    for (int j = 0; j < 8; j++) {
      size_t r0 = (size_t)(rb + j) * N, r1 = (size_t)(rb + j + 16) * N;
      C[r0 + col0] = c00[j];
      C[r0 + col1] = c01[j];
      C[r1 + col0] = c10[j];
      C[r1 + col1] = c11[j];
    }
  }
}

// ----------------- Combine: out += sum_j p_j * exp_out[slot_j] -------------
__global__ __launch_bounds__(256) void combine_kernel(
    const float* __restrict__ exp_out, const float* __restrict__ top_p,
    const int* __restrict__ top_i, const int* __restrict__ rank,
    float* __restrict__ out) {
  int t = blockIdx.x;
  int s0 = -1, s1 = -1;
  float p0 = 0.f, p1 = 0.f;
  {
    int r = rank[t * 2];
    if (r < CAP) { s0 = top_i[t * 2] * CAP + r; p0 = top_p[t * 2]; }
    r = rank[t * 2 + 1];
    if (r < CAP) { s1 = top_i[t * 2 + 1] * CAP + r; p1 = top_p[t * 2 + 1]; }
  }
#pragma unroll
  for (int c = 0; c < 4; c++) {
    int hcol = threadIdx.x + c * 256;
    float v = out[(size_t)t * HIDDEN + hcol];
    if (s0 >= 0) v += exp_out[(size_t)s0 * HIDDEN + hcol] * p0;
    if (s1 >= 0) v += exp_out[(size_t)s1 * HIDDEN + hcol] * p1;
    out[(size_t)t * HIDDEN + hcol] = v;
  }
}

// ---------------------------------------------------------------------------
extern "C" void kernel_launch(void* const* d_in, const int* in_sizes, int n_in,
                              void* d_out, int out_size, void* d_ws,
                              size_t ws_size, hipStream_t stream) {
  (void)in_sizes; (void)n_in; (void)out_size; (void)ws_size;
  const float* hidden = (const float*)d_in[0];
  const float* ln1w = (const float*)d_in[1];
  const float* ln1b = (const float*)d_in[2];
  const float* ln2w = (const float*)d_in[3];
  const float* ln2b = (const float*)d_in[4];
  const float* qkvW = (const float*)d_in[5];
  const float* projW = (const float*)d_in[6];
  const float* routerW = (const float*)d_in[7];
  const float* w1 = (const float*)d_in[8];
  const float* w2 = (const float*)d_in[9];
  float* out = (float*)d_out;

  char* ws = (char*)d_ws;
  size_t off = 0;
  auto take = [&](size_t bytes) -> char* {
    char* p = ws + off;
    off += (bytes + 255) & ~(size_t)255;
    return p;
  };
  __bf16* ln1    = (__bf16*)take((size_t)NTOK * HIDDEN * 2);
  float*  qkv    = (float*)take((size_t)NTOK * QKV_OUT * 4);
  __bf16* attn   = (__bf16*)take((size_t)NTOK * HIDDEN * 2);
  __bf16* ln2    = (__bf16*)take((size_t)NTOK * HIDDEN * 2);
  float*  top_p  = (float*)take((size_t)NTOK * 2 * 4);
  int*    top_i  = (int*)take((size_t)NTOK * 2 * 4);
  int*    rankb  = (int*)take((size_t)NTOK * 2 * 4);
  __bf16* exp_in = (__bf16*)take((size_t)NEXP * CAP * HIDDEN * 2);
  __bf16* act    = (__bf16*)take((size_t)NEXP * CAP * FFN * 2);
  float*  expo   = (float*)take((size_t)NEXP * CAP * HIDDEN * 4);

  // 1) LN1 -> bf16
  ln_bf16_kernel<<<NTOK, 256, 0, stream>>>(hidden, ln1w, ln1b, ln1);
  // 2) QKV = ln1 x qkvW^T
  gemm_wt_kernel<0><<<dim3(NTOK / 64, QKV_OUT / 128), 256, 0, stream>>>(
      ln1, qkvW, qkv, nullptr, QKV_OUT, HIDDEN);
  // 3) causal GQA flash attention
  flash_attn_kernel<<<BATCH * NHEADS * (S_LEN / 16), 32, 0, stream>>>(qkv, attn);
  // 4) out = hidden + attn x projW^T   (residual fused)
  gemm_wt_kernel<1><<<dim3(NTOK / 64, HIDDEN / 128), 256, 0, stream>>>(
      attn, projW, out, hidden, HIDDEN, HIDDEN);
  // 5) LN2 -> bf16
  ln_bf16_kernel<<<NTOK, 256, 0, stream>>>(out, ln2w, ln2b, ln2);
  // 6) router top-2
  router_kernel<<<NTOK / 8, 256, 0, stream>>>(ln2, routerW, top_p, top_i);
  // 7) stable per-expert ranks (deterministic capacity drop)
  rank_kernel<<<1, 32, 0, stream>>>(top_i, rankb);
  // 8) padded dispatch
  dispatch_kernel<<<NTOK * 2, 256, 0, stream>>>(ln2, top_i, rankb, exp_in);
  // 9) expert FFN: act = gelu(exp_in x w1)
  gemm_moe_kernel<0><<<dim3(CAP / 64, FFN / 128, NEXP), 256, 0, stream>>>(
      exp_in, w1, (void*)act, FFN, HIDDEN, (size_t)CAP * HIDDEN,
      (size_t)HIDDEN * FFN, (size_t)CAP * FFN);
  // 10) exp_out = act x w2
  gemm_moe_kernel<1><<<dim3(CAP / 64, HIDDEN / 128, NEXP), 256, 0, stream>>>(
      act, w2, (void*)expo, HIDDEN, FFN, (size_t)CAP * FFN,
      (size_t)FFN * HIDDEN, (size_t)CAP * HIDDEN);
  // 11) out += weighted combine (no atomics; deterministic)
  combine_kernel<<<NTOK, 256, 0, stream>>>(expo, top_p, top_i, rankb, out);
}
